// EdgeLocalGATPolicy_14482629722503
// MI455X (gfx1250) — matrix operand order repaired and checked by reading, not compiled, not run on hardware
//
#include <hip/hip_runtime.h>
#include <hip/hip_bf16.h>
#include <cfloat>
#include <math.h>

typedef __attribute__((ext_vector_type(16))) __bf16 v16bf;
typedef __attribute__((ext_vector_type(8)))  __bf16 v8bf;
typedef __attribute__((ext_vector_type(8)))  float  v8f;

#define HD 128
#define NEG_BIG (-3.402823466e+38f)
#define LN_EPS 1e-5f
#define INV_SQRT_H 0.08838834764831845f   /* 1/sqrt(128) */

__device__ __forceinline__ float gelu_f(float x) {
    return 0.5f * x * (1.0f + erff(x * 0.7071067811865475f));
}

__device__ __forceinline__ float waveSum(float v) {
#pragma unroll
    for (int m = 16; m >= 1; m >>= 1) v += __shfl_xor(v, m, 32);
    return v;
}

__device__ __forceinline__ float blockSum(float v, float* s, int nw) {
    v = waveSum(v);
    int w = threadIdx.x >> 5, ln = threadIdx.x & 31;
    __syncthreads();
    if (ln == 0) s[w] = v;
    __syncthreads();
    float r = 0.f;
    for (int i = 0; i < nw; ++i) r += s[i];
    return r;
}

__device__ __forceinline__ float blockMax(float v, float* s, int nw) {
#pragma unroll
    for (int m = 16; m >= 1; m >>= 1) v = fmaxf(v, __shfl_xor(v, m, 32));
    int w = threadIdx.x >> 5, ln = threadIdx.x & 31;
    __syncthreads();
    if (ln == 0) s[w] = v;
    __syncthreads();
    float r = NEG_BIG;
    for (int i = 0; i < nw; ++i) r = fmaxf(r, s[i]);
    return r;
}

// ---------------------------------------------------------------------------
// Kernel A: q_film = gelu(ln(question)@qf_w + qf_b)
//           state_base = ln(state); q = state_base@wq; qk = q @ wk^T
// grid = B, block = 128
// ---------------------------------------------------------------------------
__global__ __launch_bounds__(128) void prelim_kernel(
    const float* __restrict__ question, const float* __restrict__ state,
    const float* __restrict__ qf_ln_g, const float* __restrict__ qf_ln_b,
    const float* __restrict__ qf_w,    const float* __restrict__ qf_b,
    const float* __restrict__ sn_g,    const float* __restrict__ sn_b,
    const float* __restrict__ wq,      const float* __restrict__ wk,
    float* __restrict__ qfilm,         float* __restrict__ qk)
{
    __shared__ float buf[HD];
    __shared__ float qv[HD];
    __shared__ float s[4];
    int b = blockIdx.x, t = threadIdx.x;

    // q_film
    float x  = question[b * HD + t];
    float mu = blockSum(x, s, 4) * (1.0f / HD);
    float d  = x - mu;
    float var = blockSum(d * d, s, 4) * (1.0f / HD);
    buf[t] = d * rsqrtf(var + LN_EPS) * qf_ln_g[t] + qf_ln_b[t];
    __syncthreads();
    float acc = 0.f;
    for (int k = 0; k < HD; ++k) acc += buf[k] * qf_w[k * HD + t];
    qfilm[b * HD + t] = gelu_f(acc + qf_b[t]);
    __syncthreads();

    // state base -> q -> qk
    x   = state[b * HD + t];
    mu  = blockSum(x, s, 4) * (1.0f / HD);
    d   = x - mu;
    var = blockSum(d * d, s, 4) * (1.0f / HD);
    buf[t] = d * rsqrtf(var + LN_EPS) * sn_g[t] + sn_b[t];
    __syncthreads();
    acc = 0.f;
    for (int k = 0; k < HD; ++k) acc += buf[k] * wq[k * HD + t];
    qv[t] = acc;
    __syncthreads();
    acc = 0.f;
    for (int n = 0; n < HD; ++n) acc += qv[n] * wk[t * HD + n];
    qk[b * HD + t] = acc;
}

// ---------------------------------------------------------------------------
// Kernel B: fused  er = gelu(ln(edge + dir_emb + q_film[seg]) @ ep_w + ep_b)
//           att = (er . qk[seg]) / sqrt(H)  (masked), er stored bf16.
// block = 128 (4 waves), each wave owns a 16-edge tile; grid = E/64
// ---------------------------------------------------------------------------
__global__ __launch_bounds__(128) void edge_kernel(
    const float* __restrict__ et,   const int* __restrict__ ebatch,
    const unsigned char* __restrict__ selm, const unsigned char* __restrict__ frm,
    const unsigned char* __restrict__ valm, const int* __restrict__ edir,
    const float* __restrict__ dir_emb, int dirv,
    const float* __restrict__ qfilm,  const float* __restrict__ qk,
    const float* __restrict__ ep_ln_g, const float* __restrict__ ep_ln_b,
    const float* __restrict__ ep_w,   const float* __restrict__ ep_b,
    __bf16* __restrict__ er_out, float* __restrict__ att_out,
    unsigned char* __restrict__ cand_out, int E)
{
    __shared__ __bf16 WtB[HD][HD + 8];        // ep_w transposed: [n][k], padded rows
    __shared__ __bf16 stage[4][16][HD + 8];   // normalized A rows per wave

    int t = threadIdx.x;
    int wave = t >> 5, lane = t & 31, ln16 = lane & 15, hi = lane >> 4;

    for (int k = 0; k < HD; ++k) WtB[t][k] = (__bf16)ep_w[k * HD + t];

    int e0 = blockIdx.x * 64 + wave * 16;
    int seg8[8];
    const int c4 = lane * 4;
    float g4[4], b4[4];
#pragma unroll
    for (int j = 0; j < 4; ++j) { g4[j] = ep_ln_g[c4 + j]; b4[j] = ep_ln_b[c4 + j]; }

#pragma unroll
    for (int r = 0; r < 16; ++r) {
        int e = e0 + r;
        int seg = ebatch[e];
        int dd = edir[e]; dd = dd < 0 ? 0 : (dd >= dirv ? dirv - 1 : dd);
        const float* etr = et      + (size_t)e   * HD;
        const float* der = dir_emb + (size_t)dd  * HD;
        const float* qfr = qfilm   + (size_t)seg * HD;
        float xv[4], s1 = 0.f, s2 = 0.f;
#pragma unroll
        for (int j = 0; j < 4; ++j) {
            float x = etr[c4 + j] + der[c4 + j] + qfr[c4 + j];
            xv[j] = x; s1 += x; s2 += x * x;
        }
        s1 = waveSum(s1); s2 = waveSum(s2);
        float mu   = s1 * (1.0f / HD);
        float var  = s2 * (1.0f / HD) - mu * mu;
        float rstd = rsqrtf(var + LN_EPS);
#pragma unroll
        for (int j = 0; j < 4; ++j)
            stage[wave][r][c4 + j] = (__bf16)((xv[j] - mu) * rstd * g4[j] + b4[j]);
        if ((r >> 3) == hi) seg8[r & 7] = seg;
    }
    __syncthreads();

    v8f acc[8];
    const v8f vzero = {0.f, 0.f, 0.f, 0.f, 0.f, 0.f, 0.f, 0.f};
#pragma unroll
    for (int nt = 0; nt < 8; ++nt) acc[nt] = vzero;

#pragma unroll
    for (int kk = 0; kk < 4; ++kk) {
        int base = kk * 32 + hi * 8;
        const __bf16* ar = &stage[wave][ln16][0];
        v8bf alo = *(const v8bf*)(ar + base);
        v8bf ahi = *(const v8bf*)(ar + base + 16);
        v16bf afrag = __builtin_shufflevector(alo, ahi,
            0, 1, 2, 3, 4, 5, 6, 7, 8, 9, 10, 11, 12, 13, 14, 15);
#pragma unroll
        for (int nt = 0; nt < 8; ++nt) {
            const __bf16* br = &WtB[nt * 16 + ln16][kk * 32 + hi * 8];
            v8bf blo = *(const v8bf*)br;
            v8bf bhi = *(const v8bf*)(br + 16);
            v16bf bfrag = __builtin_shufflevector(blo, bhi,
                0, 1, 2, 3, 4, 5, 6, 7, 8, 9, 10, 11, 12, 13, 14, 15);
            acc[nt] = __builtin_amdgcn_wmma_f32_16x16x32_bf16(
                false, afrag, false, bfrag, (short)0, acc[nt], false, false);
        }
    }

    // epilogue: bias + exact-erf gelu, store er (bf16), att partials
    float p[8] = {0.f, 0.f, 0.f, 0.f, 0.f, 0.f, 0.f, 0.f};
#pragma unroll
    for (int nt = 0; nt < 8; ++nt) {
        int c = nt * 16 + ln16;
        float bias = ep_b[c];
#pragma unroll
        for (int i = 0; i < 8; ++i) {
            float v = gelu_f(acc[nt][i] + bias);
            int row = hi * 8 + i;
            er_out[(size_t)(e0 + row) * HD + c] = (__bf16)v;
            p[i] += v * qk[(size_t)seg8[i] * HD + c];
        }
    }
#pragma unroll
    for (int i = 0; i < 8; ++i) {
#pragma unroll
        for (int m = 8; m >= 1; m >>= 1) p[i] += __shfl_xor(p[i], m, 32);
    }
    if (ln16 < 8) {
        float attv = 0.f;
#pragma unroll
        for (int i = 0; i < 8; ++i) if (ln16 == i) attv = p[i];
        int e = e0 + hi * 8 + ln16;
        bool c = (!selm[e]) && frm[e] && valm[e];
        cand_out[e] = c ? 1 : 0;
        att_out[e]  = c ? attv * INV_SQRT_H : NEG_BIG;
    }
}

// ---------------------------------------------------------------------------
// Kernel C: per-batch segment softmax + ctx_pre = sum(att_w * er)  [B,H]
// grid = B, block = 256 (edge_batch is sorted -> binary search the segment)
// ---------------------------------------------------------------------------
__global__ __launch_bounds__(256) void ctx_kernel(
    const float* __restrict__ att, const unsigned char* __restrict__ cand,
    const __bf16* __restrict__ er, const int* __restrict__ ebatch,
    float* __restrict__ ctx_pre, int E)
{
    __shared__ float s[8];
    __shared__ float halfacc[HD];
    int b = blockIdx.x, t = threadIdx.x;

    int lo, hiN;
    { int l = 0, r = E; while (l < r) { int m = (l + r) >> 1; if (ebatch[m] < b) l = m + 1; else r = m; } lo = l; }
    { int l = lo, r = E; while (l < r) { int m = (l + r) >> 1; if (ebatch[m] < b + 1) l = m + 1; else r = m; } hiN = l; }

    float m = NEG_BIG;
    for (int e = lo + t; e < hiN; e += 256) m = fmaxf(m, att[e]);
    m = blockMax(m, s, 8);

    float sum = 0.f;
    for (int e = lo + t; e < hiN; e += 256)
        if (cand[e]) sum += expf(att[e] - m);
    sum = blockSum(sum, s, 8);
    float wscale = 1.0f / fmaxf(sum, 1.1920928955078125e-7f);  // FLT_EPSILON

    int col = t & (HD - 1), half = t >> 7;
    float acc = 0.f;
    for (int e = lo + half; e < hiN; e += 2)
        if (cand[e]) acc += expf(att[e] - m) * (float)er[(size_t)e * HD + col];
    if (half) halfacc[col] = acc;
    __syncthreads();
    if (!half) ctx_pre[b * HD + col] = (acc + halfacc[col]) * wscale;
}

// ---------------------------------------------------------------------------
// Kernel D: context = ctx_pre@wv ; state_out = ln(state+context) ; stop head
// grid = B, block = 128
// ---------------------------------------------------------------------------
__global__ __launch_bounds__(128) void finish_kernel(
    const float* __restrict__ ctxp, const float* __restrict__ wv,
    const float* __restrict__ state,
    const float* __restrict__ sn_g, const float* __restrict__ sn_b,
    const float* __restrict__ sh_ln_g, const float* __restrict__ sh_ln_b,
    const float* __restrict__ sh_w1, const float* __restrict__ sh_b1,
    const float* __restrict__ sh_w2, const float* __restrict__ sh_b2,
    float* __restrict__ out_state, float* __restrict__ out_stop)
{
    __shared__ float pre[HD];
    __shared__ float cc[2 * HD];
    __shared__ float s[4];
    int b = blockIdx.x, t = threadIdx.x;

    pre[t] = ctxp[b * HD + t];
    __syncthreads();
    float acc = 0.f;
    for (int n = 0; n < HD; ++n) acc += pre[n] * wv[n * HD + t];

    float st = state[b * HD + t];
    float x  = st + acc;
    float mu = blockSum(x, s, 4) * (1.0f / HD);
    float d  = x - mu;
    float var = blockSum(d * d, s, 4) * (1.0f / HD);
    float so  = d * rsqrtf(var + LN_EPS) * sn_g[t] + sn_b[t];
    out_state[b * HD + t] = so;

    // stop head: ln over concat(state_out, state) (256)
    float mu2  = blockSum(so + st, s, 4) * (1.0f / 256.0f);
    float d1 = so - mu2, d2 = st - mu2;
    float var2 = blockSum(d1 * d1 + d2 * d2, s, 4) * (1.0f / 256.0f);
    float rs2  = rsqrtf(var2 + LN_EPS);
    cc[t]      = d1 * rs2 * sh_ln_g[t]      + sh_ln_b[t];
    cc[HD + t] = d2 * rs2 * sh_ln_g[HD + t] + sh_ln_b[HD + t];
    __syncthreads();
    float h = 0.f;
    for (int k = 0; k < 2 * HD; ++k) h += cc[k] * sh_w1[k * HD + t];
    h = gelu_f(h + sh_b1[t]);
    float tot = blockSum(h * sh_w2[t], s, 4);
    if (t == 0) out_stop[b] = tot + sh_b2[0];
}

// ---------------------------------------------------------------------------
// Kernel E: edge head.  h = gelu(ln(concat(state_out[seg], er)) @ eh_w1 + b1)
//           logits = h@eh_w2 + b2 (masked). K=256 done in two 128-wide phases.
// block = 128 (4 waves x 16-edge tiles); grid = E/64
// ---------------------------------------------------------------------------
__global__ __launch_bounds__(128) void edgehead_kernel(
    const float* __restrict__ state_out, const __bf16* __restrict__ er,
    const int* __restrict__ ebatch, const unsigned char* __restrict__ cand,
    const float* __restrict__ eh_ln_g, const float* __restrict__ eh_ln_b,
    const float* __restrict__ eh_w1,   const float* __restrict__ eh_b1,
    const float* __restrict__ eh_w2,   const float* __restrict__ eh_b2,
    float* __restrict__ logits, int E)
{
    __shared__ __bf16 Wt1[HD][HD + 8];   // one 128-K phase of eh_w1, [n][k]

    int t = threadIdx.x;
    int wave = t >> 5, lane = t & 31, ln16 = lane & 15, hi = lane >> 4;
    int e0 = blockIdx.x * 64 + wave * 16;

    // per-row LN stats over the 256-wide concat; capture my A-row's stats
    float meanMy = 0.f, rstdMy = 0.f; int segMy = 0;
#pragma unroll
    for (int r = 0; r < 16; ++r) {
        int e = e0 + r;
        int seg = ebatch[e];
        float s1 = 0.f, s2 = 0.f;
        if (lane < 16) {
            const float* sp = state_out + (size_t)seg * HD + lane * 8;
#pragma unroll
            for (int j = 0; j < 8; ++j) { float v = sp[j]; s1 += v; s2 += v * v; }
        } else {
            const __bf16* ep = er + (size_t)e * HD + (lane - 16) * 8;
#pragma unroll
            for (int j = 0; j < 8; ++j) { float v = (float)ep[j]; s1 += v; s2 += v * v; }
        }
        s1 = waveSum(s1); s2 = waveSum(s2);
        float mu   = s1 * (1.0f / 256.0f);
        float var  = s2 * (1.0f / 256.0f) - mu * mu;
        float rstd = rsqrtf(var + LN_EPS);
        if (ln16 == r) { meanMy = mu; rstdMy = rstd; segMy = seg; }
    }

    v8f acc[8];
    const v8f vzero = {0.f, 0.f, 0.f, 0.f, 0.f, 0.f, 0.f, 0.f};
#pragma unroll
    for (int nt = 0; nt < 8; ++nt) acc[nt] = vzero;

#pragma unroll
    for (int p = 0; p < 2; ++p) {
        __syncthreads();
        for (int k = 0; k < HD; ++k)
            Wt1[t][k] = (__bf16)eh_w1[(size_t)(p * HD + k) * HD + t];
        __syncthreads();
#pragma unroll
        for (int kk = 0; kk < 4; ++kk) {
            int kkg = p * 4 + kk;
            int e = e0 + ln16;
            v16bf afrag;
#pragma unroll
            for (int ei = 0; ei < 16; ++ei) {
                int c = kkg * 32 + hi * 8 + (ei < 8 ? ei : ei + 8);
                float v;
                if (p == 0) v = state_out[(size_t)segMy * HD + c];
                else        v = (float)er[(size_t)e * HD + (c - HD)];
                v = (v - meanMy) * rstdMy * eh_ln_g[c] + eh_ln_b[c];
                afrag[ei] = (__bf16)v;
            }
#pragma unroll
            for (int nt = 0; nt < 8; ++nt) {
                const __bf16* br = &Wt1[nt * 16 + ln16][kk * 32 + hi * 8];
                v8bf blo = *(const v8bf*)br;
                v8bf bhi = *(const v8bf*)(br + 16);
                v16bf bfrag = __builtin_shufflevector(blo, bhi,
                    0, 1, 2, 3, 4, 5, 6, 7, 8, 9, 10, 11, 12, 13, 14, 15);
                acc[nt] = __builtin_amdgcn_wmma_f32_16x16x32_bf16(
                    false, afrag, false, bfrag, (short)0, acc[nt], false, false);
            }
        }
    }

    float pv[8] = {0.f, 0.f, 0.f, 0.f, 0.f, 0.f, 0.f, 0.f};
#pragma unroll
    for (int nt = 0; nt < 8; ++nt) {
        int c = nt * 16 + ln16;
        float bias = eh_b1[c];
        float w2   = eh_w2[c];
#pragma unroll
        for (int i = 0; i < 8; ++i)
            pv[i] += gelu_f(acc[nt][i] + bias) * w2;
    }
#pragma unroll
    for (int i = 0; i < 8; ++i) {
#pragma unroll
        for (int m = 8; m >= 1; m >>= 1) pv[i] += __shfl_xor(pv[i], m, 32);
    }
    if (ln16 < 8) {
        float lv = 0.f;
#pragma unroll
        for (int i = 0; i < 8; ++i) if (ln16 == i) lv = pv[i];
        int e = e0 + hi * 8 + ln16;
        logits[e] = cand[e] ? (lv + eh_b2[0]) : NEG_BIG;
    }
}

// ---------------------------------------------------------------------------
extern "C" void kernel_launch(void* const* d_in, const int* in_sizes, int n_in,
                              void* d_out, int out_size, void* d_ws, size_t ws_size,
                              hipStream_t stream)
{
    (void)n_in; (void)out_size; (void)ws_size;

    const float* edge_tokens = (const float*)d_in[0];
    const float* question    = (const float*)d_in[1];
    const float* state       = (const float*)d_in[2];
    const int*   ebatch      = (const int*)d_in[3];
    const unsigned char* selm = (const unsigned char*)d_in[4];
    const unsigned char* frm  = (const unsigned char*)d_in[5];
    const unsigned char* valm = (const unsigned char*)d_in[6];
    const int*   edir        = (const int*)d_in[7];
    const float* sn_g    = (const float*)d_in[8];
    const float* sn_b    = (const float*)d_in[9];
    const float* qf_ln_g = (const float*)d_in[10];
    const float* qf_ln_b = (const float*)d_in[11];
    const float* qf_w    = (const float*)d_in[12];
    const float* qf_b    = (const float*)d_in[13];
    const float* dir_emb = (const float*)d_in[14];
    const float* ep_ln_g = (const float*)d_in[15];
    const float* ep_ln_b = (const float*)d_in[16];
    const float* ep_w    = (const float*)d_in[17];
    const float* ep_b    = (const float*)d_in[18];
    const float* wq      = (const float*)d_in[19];
    const float* wk      = (const float*)d_in[20];
    const float* wv      = (const float*)d_in[21];
    const float* eh_ln_g = (const float*)d_in[22];
    const float* eh_ln_b = (const float*)d_in[23];
    const float* eh_w1   = (const float*)d_in[24];
    const float* eh_b1   = (const float*)d_in[25];
    const float* eh_w2   = (const float*)d_in[26];
    const float* eh_b2   = (const float*)d_in[27];
    const float* sh_ln_g = (const float*)d_in[28];
    const float* sh_ln_b = (const float*)d_in[29];
    const float* sh_w1   = (const float*)d_in[30];
    const float* sh_b1   = (const float*)d_in[31];
    const float* sh_w2   = (const float*)d_in[32];
    const float* sh_b2   = (const float*)d_in[33];

    const int E = in_sizes[0] / HD;      // 400000
    const int B = in_sizes[1] / HD;      // 256
    const int dirv = in_sizes[14] / HD;  // 3

    // workspace carve-up
    char* w = (char*)d_ws;
    __bf16* er   = (__bf16*)w;          w += (size_t)E * HD * sizeof(__bf16);
    float*  att  = (float*)w;           w += (size_t)E * sizeof(float);
    unsigned char* cand = (unsigned char*)w;
    w += (((size_t)E + 255) & ~(size_t)255);
    float* qfilm = (float*)w;           w += (size_t)B * HD * sizeof(float);
    float* qk    = (float*)w;           w += (size_t)B * HD * sizeof(float);
    float* ctxp  = (float*)w;           w += (size_t)B * HD * sizeof(float);

    float* out_logits = (float*)d_out;
    float* out_stop   = out_logits + E;
    float* out_state  = out_stop + B;

    // A: small per-graph precompute (incl. qk = q @ wk^T folding)
    prelim_kernel<<<B, 128, 0, stream>>>(question, state, qf_ln_g, qf_ln_b,
                                         qf_w, qf_b, sn_g, sn_b, wq, wk,
                                         qfilm, qk);
    // B: fused edge projection + attention scores (WMMA bf16)
    edge_kernel<<<E / 64, 128, 0, stream>>>(edge_tokens, ebatch, selm, frm, valm,
                                            edir, dir_emb, dirv, qfilm, qk,
                                            ep_ln_g, ep_ln_b, ep_w, ep_b,
                                            er, att, cand, E);
    // C: segment softmax + weighted er sum
    ctx_kernel<<<B, 256, 0, stream>>>(att, cand, er, ebatch, ctxp, E);
    // D: context@wv, state_out, stop head
    finish_kernel<<<B, 128, 0, stream>>>(ctxp, wv, state, sn_g, sn_b,
                                         sh_ln_g, sh_ln_b, sh_w1, sh_b1,
                                         sh_w2, sh_b2, out_state, out_stop);
    // E: edge scoring head (WMMA bf16, K=256 in two phases)
    edgehead_kernel<<<E / 64, 128, 0, stream>>>(out_state, er, ebatch, cand,
                                                eh_ln_g, eh_ln_b, eh_w1, eh_b1,
                                                eh_w2, eh_b2, out_logits, E);
}
